// TSMLRTAS_32040456028907
// MI455X (gfx1250) — matrix operand order repaired and checked by reading, not compiled, not run on hardware
//
#include <hip/hip_runtime.h>

#define D_IN    2048
#define D_MODEL 64
#define D_STATE 128
#define STAGES  20
#define N_CLS   202
#define N_CLS_PAD 208
#define BATCH   8
#define TLEN    2048
#define ROWS    (BATCH * TLEN)   // 16384

typedef __attribute__((ext_vector_type(16))) __bf16 v16bf;
typedef __attribute__((ext_vector_type(8)))  float  v8f;

union BF16x16 { unsigned u[8]; v16bf v; };

__device__ __forceinline__ unsigned short f2bf(float f) {
  unsigned u = __float_as_uint(f);
  u += 0x7FFFu + ((u >> 16) & 1u);        // round-to-nearest-even
  return (unsigned short)(u >> 16);
}
__device__ __forceinline__ unsigned pack_bf(float a, float b) {
  return (unsigned)f2bf(a) | ((unsigned)f2bf(b) << 16);
}
__device__ __forceinline__ float bf2f(unsigned short h) {
  return __uint_as_float(((unsigned)h) << 16);
}

// A fragment (16x32 bf16, MxK) from LDS tile, per CDNA5 16-bit A layout:
// lanes 0-15 = M rows, VGPR0-3: K 0..7 (lanes>=16: K 8..15), VGPR4-7: K 16..23 (24..31)
__device__ __forceinline__ v16bf load_a_frag(const unsigned* lds, int rowStrideDw,
                                             int lane, int ktOffDw) {
  int row  = lane & 15;
  int hi   = (lane >> 4) & 1;
  int base = row * rowStrideDw + ktOffDw + hi * 4;
  uint4 lo = *(const uint4*)(lds + base);
  uint4 up = *(const uint4*)(lds + base + 8);
  BF16x16 fr;
  fr.u[0] = lo.x; fr.u[1] = lo.y; fr.u[2] = lo.z; fr.u[3] = lo.w;
  fr.u[4] = up.x; fr.u[5] = up.y; fr.u[6] = up.z; fr.u[7] = up.w;
  return fr.v;
}

// B fragment: pre-swizzled in global memory, 8 contiguous dwords per lane (32B aligned)
__device__ __forceinline__ v16bf load_b_frag(const unsigned* g) {
  uint4 lo = *(const uint4*)g;
  uint4 up = *(const uint4*)(g + 4);
  BF16x16 fr;
  fr.u[0] = lo.x; fr.u[1] = lo.y; fr.u[2] = lo.z; fr.u[3] = lo.w;
  fr.u[4] = up.x; fr.u[5] = up.y; fr.u[6] = up.z; fr.u[7] = up.w;
  return fr.v;
}

__device__ __forceinline__ v8f wmma_bf16(v16bf a, v16bf b, v8f c) {
  return __builtin_amdgcn_wmma_f32_16x16x32_bf16(false, a, false, b, (short)0, c,
                                                 false, false);
}

// ---------------------------------------------------------------------------
// Pre-swizzle a (K x N) weight matrix into per-lane WMMA B-fragment layout:
// dst dword index = ((kt*ntiles + nt)*32 + lane)*8 + j
// lane n = nt*16 + lane%16 ; K pair = kt*32 + (j>=4?16:0) + (lane>=16?8:0) + 2(j&3) (+1)
// transpose==0: src[k*ld + n] ; transpose==1: src[n*ld + k]
// ---------------------------------------------------------------------------
__global__ void prep_frag(const float* __restrict__ src, unsigned* __restrict__ dst,
                          int K, int Npad, int Nsrc, int transpose, int ld,
                          int nstages, long srcStride, long dstStrideDw) {
  const int dwPer = (K * Npad) >> 1;
  long gid = (long)blockIdx.x * 256 + threadIdx.x;
  if (gid >= (long)nstages * dwPer) return;
  int s = (int)(gid / dwPer);
  int r = (int)(gid % dwPer);
  int ntiles = Npad >> 4;
  int f    = r >> 8;
  int lane = (r >> 3) & 31;
  int j    = r & 7;
  int kt = f / ntiles, nt = f % ntiles;
  int n  = nt * 16 + (lane & 15);
  int k0 = kt * 32 + ((j & 4) ? 16 : 0) + ((lane & 16) ? 8 : 0) + 2 * (j & 3);
  unsigned p = 0;
  for (int hh = 0; hh < 2; ++hh) {
    int k = k0 + hh;
    float v = 0.f;
    if (n < Nsrc)
      v = transpose ? src[(long)s * srcStride + (long)n * ld + k]
                    : src[(long)s * srcStride + (long)k * ld + n];
    p |= (unsigned)f2bf(v) << (16 * hh);
  }
  dst[(long)s * dstStrideDw + r] = p;
}

// ---------------------------------------------------------------------------
// Kernel 1: LayerNorm(D_IN) + embed GEMM (16 rows x 2048) @ (2048 x 64)
// block = 128 threads (4 waves), each wave owns one 16-wide N tile.
// ---------------------------------------------------------------------------
__global__ __launch_bounds__(128) void k_embed(
    const float* __restrict__ x, const float* __restrict__ g,
    const float* __restrict__ b, const unsigned* __restrict__ wf,
    const float* __restrict__ bias, float* __restrict__ h) {
  __shared__ unsigned lds[16 * 256];   // 16 rows x 512 bf16 chunk (16 KB)
  __shared__ float s_mean[16], s_rstd[16];
  const int tid = threadIdx.x, lane = tid & 31, wave = tid >> 5;
  const long row0 = (long)blockIdx.x * 16;

  // LN stats: wave covers 4 rows; 8 lanes/row, 256 contiguous floats per lane
  {
    int subrow = lane >> 3, sub = lane & 7;
    int r = wave * 4 + subrow;
    const float* xr = x + (row0 + r) * D_IN + sub * 256;
    float s = 0.f, sq = 0.f;
    for (int i = 0; i < 256; i += 4) {
      float4 v = *(const float4*)(xr + i);
      s  += v.x + v.y + v.z + v.w;
      sq += v.x * v.x + v.y * v.y + v.z * v.z + v.w * v.w;
    }
    s  += __shfl_xor(s, 1, 32);  s  += __shfl_xor(s, 2, 32);  s  += __shfl_xor(s, 4, 32);
    sq += __shfl_xor(sq, 1, 32); sq += __shfl_xor(sq, 2, 32); sq += __shfl_xor(sq, 4, 32);
    if (sub == 0) {
      float mean = s * (1.f / D_IN);
      float var  = sq * (1.f / D_IN) - mean * mean;
      s_mean[r] = mean;
      s_rstd[r] = rsqrtf(var + 1e-5f);
    }
  }
  __syncthreads();

  v8f acc = {};
  for (int kc = 0; kc < 4; ++kc) {
    // prefetch next K chunk (emits global_prefetch_b8)
    if (kc < 3)
      __builtin_prefetch(x + (row0 + (tid & 15)) * D_IN + (kc + 1) * 512 + (tid >> 4) * 64, 0, 0);
    // stage normalized bf16 chunk: 16 x 512
    for (int d = tid; d < 4096; d += 128) {
      int r  = d >> 8;
      int c2 = d & 255;
      int c  = kc * 512 + 2 * c2;
      float2 v = *(const float2*)(x + (row0 + r) * D_IN + c);
      float m = s_mean[r], rs = s_rstd[r];
      lds[r * 256 + c2] = pack_bf((v.x - m) * rs * g[c] + b[c],
                                  (v.y - m) * rs * g[c + 1] + b[c + 1]);
    }
    __syncthreads();
    for (int kt = 0; kt < 16; ++kt) {
      v16bf a  = load_a_frag(lds, 256, lane, kt * 16);
      int ktg  = kc * 16 + kt;
      v16bf bf = load_b_frag(wf + ((size_t)(ktg * 4 + wave) * 32 + lane) * 8);
      acc = wmma_bf16(a, bf, acc);
    }
    __syncthreads();
  }
  int n  = wave * 16 + (lane & 15);
  int hi = (lane >> 4) & 1;
  float bn = bias[n];
  for (int r = 0; r < 8; ++r)
    h[(row0 + r + hi * 8) * D_MODEL + n] = acc[r] + bn;
}

// ---------------------------------------------------------------------------
// Kernel 2 (per stage): z = LN(h); u = gamma * (B z)  [re + im], z saved (bf16)
// 16 column tiles (8 re + 8 im); each of 4 waves does 4 of them; K=64.
// ---------------------------------------------------------------------------
__global__ __launch_bounds__(128) void k_stage_pre(
    const float* __restrict__ h, const float* __restrict__ lng,
    const float* __restrict__ lnb, const unsigned* __restrict__ wfBre,
    const unsigned* __restrict__ wfBim, const float* __restrict__ gamma_log,
    float* __restrict__ u_re, float* __restrict__ u_im,
    unsigned* __restrict__ z_bf) {
  __shared__ unsigned zl[16 * 32];   // 16 x 64 bf16
  const int tid = threadIdx.x, lane = tid & 31, wave = tid >> 5;
  const long row0 = (long)blockIdx.x * 16;
  {
    int subrow = lane >> 3, sub = lane & 7;
    int r = wave * 4 + subrow;
    const float* hr = h + (row0 + r) * D_MODEL + sub * 8;
    float v[8]; float s = 0.f, sq = 0.f;
    for (int i = 0; i < 8; ++i) { v[i] = hr[i]; s += v[i]; sq += v[i] * v[i]; }
    s  += __shfl_xor(s, 1, 32);  s  += __shfl_xor(s, 2, 32);  s  += __shfl_xor(s, 4, 32);
    sq += __shfl_xor(sq, 1, 32); sq += __shfl_xor(sq, 2, 32); sq += __shfl_xor(sq, 4, 32);
    float mean = s * (1.f / 64.f);
    float rstd = rsqrtf(sq * (1.f / 64.f) - mean * mean + 1e-5f);
    for (int i = 0; i < 4; ++i) {
      int c = sub * 8 + 2 * i;
      unsigned p = pack_bf((v[2 * i] - mean) * rstd * lng[c] + lnb[c],
                           (v[2 * i + 1] - mean) * rstd * lng[c + 1] + lnb[c + 1]);
      zl[r * 32 + sub * 4 + i] = p;
      z_bf[(row0 + r) * 32 + sub * 4 + i] = p;
    }
  }
  __syncthreads();
  const int hi = (lane >> 4) & 1;
  v16bf a0 = load_a_frag(zl, 32, lane, 0);
  v16bf a1 = load_a_frag(zl, 32, lane, 16);
  for (int i = 0; i < 4; ++i) {
    int c    = wave + 4 * i;        // 0..15 (re: 0-7, im: 8-15)
    int isim = c >> 3;
    int nt   = c & 7;
    const unsigned* wf = isim ? wfBim : wfBre;
    v8f acc = {};
    acc = wmma_bf16(a0, load_b_frag(wf + ((size_t)(0 * 8 + nt) * 32 + lane) * 8), acc);
    acc = wmma_bf16(a1, load_b_frag(wf + ((size_t)(1 * 8 + nt) * 32 + lane) * 8), acc);
    int n = nt * 16 + (lane & 15);
    float gam = expf(gamma_log[n]);
    float* dst = isim ? u_im : u_re;
    for (int r = 0; r < 8; ++r)
      dst[(row0 + r + hi * 8) * D_STATE + n] = acc[r] * gam;
  }
}

// ---------------------------------------------------------------------------
// Kernel 3 (per stage): complex diagonal scan h_t = lam*h_{t-1} + u_t (in place).
// One block per (batch, state) channel; 128 threads x 16-step chunks,
// Hillis-Steele scan over chunk-level affine maps in LDS, then replay.
// ---------------------------------------------------------------------------
__global__ __launch_bounds__(128) void k_scan(
    const float* __restrict__ nu_log, const float* __restrict__ theta_log,
    float* __restrict__ u_re, float* __restrict__ u_im) {
  __shared__ float sAre[128], sAim[128], sHre[128], sHim[128];
  const int j  = threadIdx.x;
  const int ch = blockIdx.x;
  const int bb = ch >> 7;
  const int n  = ch & 127;
  float mag = expf(-expf(nu_log[n]));
  float th  = expf(theta_log[n]);
  float lre = mag * cosf(th);
  float lim = mag * sinf(th);

  const long base = ((long)bb * TLEN + (long)j * 16) * D_STATE + n;
  float ur[16], ui[16];
  float are = 1.f, aim = 0.f, hre = 0.f, him = 0.f;
  for (int k = 0; k < 16; ++k) {
    long idx = base + (long)k * D_STATE;
    ur[k] = u_re[idx]; ui[k] = u_im[idx];
    float tre = lre * hre - lim * him + ur[k];
    float tim = lre * him + lim * hre + ui[k];
    hre = tre; him = tim;
    float nre = lre * are - lim * aim;
    float nim = lre * aim + lim * are;
    are = nre; aim = nim;
  }
  sAre[j] = are; sAim[j] = aim; sHre[j] = hre; sHim[j] = him;
  for (int off = 1; off < 128; off <<= 1) {
    __syncthreads();
    float pare = 1.f, paim = 0.f, phre = 0.f, phim = 0.f;
    if (j >= off) { pare = sAre[j - off]; paim = sAim[j - off];
                    phre = sHre[j - off]; phim = sHim[j - off]; }
    __syncthreads();
    float nhre = are * phre - aim * phim + hre;
    float nhim = are * phim + aim * phre + him;
    float nare = are * pare - aim * paim;
    float naim = are * paim + aim * pare;
    hre = nhre; him = nhim; are = nare; aim = naim;
    sAre[j] = are; sAim[j] = aim; sHre[j] = hre; sHim[j] = him;
  }
  __syncthreads();
  float pre = 0.f, pim = 0.f;
  if (j > 0) { pre = sHre[j - 1]; pim = sHim[j - 1]; }
  for (int k = 0; k < 16; ++k) {
    float tre = lre * pre - lim * pim + ur[k];
    float tim = lre * pim + lim * pre + ui[k];
    pre = tre; pim = tim;
    long idx = base + (long)k * D_STATE;
    u_re[idx] = pre; u_im[idx] = pim;
  }
}

// ---------------------------------------------------------------------------
// Kernel 4 (per stage): y = h_re C_re - h_im C_im + Dskip*z ; h += y.  K=128.
// ---------------------------------------------------------------------------
__global__ __launch_bounds__(128) void k_stage_post(
    const float* __restrict__ u_re, const float* __restrict__ u_im,
    const unsigned* __restrict__ wfCre, const unsigned* __restrict__ wfCim,
    const float* __restrict__ dskip, const unsigned* __restrict__ z_bf,
    float* __restrict__ h) {
  __shared__ unsigned are_l[16 * 64];   // 16 x 128 bf16
  __shared__ unsigned aim_l[16 * 64];   // negated imag part
  const int tid = threadIdx.x, lane = tid & 31, wave = tid >> 5;
  const long row0 = (long)blockIdx.x * 16;
  for (int d = tid; d < 1024; d += 128) {
    int r = d >> 6, c2 = d & 63;
    long src = (row0 + r) * D_STATE + 2 * c2;
    float2 vr = *(const float2*)(u_re + src);
    float2 vi = *(const float2*)(u_im + src);
    are_l[d] = pack_bf(vr.x, vr.y);
    aim_l[d] = pack_bf(-vi.x, -vi.y);
  }
  __syncthreads();
  v8f acc = {};
  for (int kt = 0; kt < 4; ++kt) {
    v16bf ar = load_a_frag(are_l, 64, lane, kt * 16);
    acc = wmma_bf16(ar, load_b_frag(wfCre + ((size_t)(kt * 4 + wave) * 32 + lane) * 8), acc);
    v16bf ai = load_a_frag(aim_l, 64, lane, kt * 16);
    acc = wmma_bf16(ai, load_b_frag(wfCim + ((size_t)(kt * 4 + wave) * 32 + lane) * 8), acc);
  }
  int n  = wave * 16 + (lane & 15);
  int hi = (lane >> 4) & 1;
  float dn = dskip[n];
  for (int r = 0; r < 8; ++r) {
    long row = row0 + r + hi * 8;
    unsigned zp = z_bf[row * 32 + (n >> 1)];
    float zf = bf2f((unsigned short)((n & 1) ? (zp >> 16) : (zp & 0xFFFFu)));
    h[row * D_MODEL + n] += acc[r] + dn * zf;
  }
}

// ---------------------------------------------------------------------------
// Kernel 5: classifier GEMM (16 x 64) @ (64 x 208 padded), store n < 202.
// ---------------------------------------------------------------------------
__global__ __launch_bounds__(128) void k_cls(
    const float* __restrict__ h, const unsigned* __restrict__ wf,
    const float* __restrict__ bc, float* __restrict__ out) {
  __shared__ unsigned hl[16 * 32];
  const int tid = threadIdx.x, lane = tid & 31, wave = tid >> 5;
  const long row0 = (long)blockIdx.x * 16;
  for (int d = tid; d < 512; d += 128) {
    int r = d >> 5, c2 = d & 31;
    float2 v = *(const float2*)(h + (row0 + r) * D_MODEL + 2 * c2);
    hl[d] = pack_bf(v.x, v.y);
  }
  __syncthreads();
  const int hi = (lane >> 4) & 1;
  v16bf a0 = load_a_frag(hl, 32, lane, 0);
  v16bf a1 = load_a_frag(hl, 32, lane, 16);
  for (int nt = wave; nt < 13; nt += 4) {
    v8f acc = {};
    acc = wmma_bf16(a0, load_b_frag(wf + ((size_t)(0 * 13 + nt) * 32 + lane) * 8), acc);
    acc = wmma_bf16(a1, load_b_frag(wf + ((size_t)(1 * 13 + nt) * 32 + lane) * 8), acc);
    int n = nt * 16 + (lane & 15);
    if (n < N_CLS) {
      float bn = bc[n];
      for (int r = 0; r < 8; ++r)
        out[(row0 + r + hi * 8) * N_CLS + n] = acc[r] + bn;
    }
  }
}

// ---------------------------------------------------------------------------
extern "C" void kernel_launch(void* const* d_in, const int* in_sizes, int n_in,
                              void* d_out, int out_size, void* d_ws, size_t ws_size,
                              hipStream_t stream) {
  (void)in_sizes; (void)n_in; (void)out_size; (void)ws_size;
  const float* x         = (const float*)d_in[0];
  const float* ln0_g     = (const float*)d_in[1];
  const float* ln0_b     = (const float*)d_in[2];
  const float* W_embed   = (const float*)d_in[3];
  const float* b_embed   = (const float*)d_in[4];
  const float* ln_g      = (const float*)d_in[5];
  const float* ln_b      = (const float*)d_in[6];
  const float* nu_log    = (const float*)d_in[7];
  const float* theta_log = (const float*)d_in[8];
  const float* gamma_log = (const float*)d_in[9];
  const float* B_re      = (const float*)d_in[10];
  const float* B_im      = (const float*)d_in[11];
  const float* C_re      = (const float*)d_in[12];
  const float* C_im      = (const float*)d_in[13];
  const float* Dskip     = (const float*)d_in[14];
  const float* W_cls     = (const float*)d_in[15];
  const float* b_cls     = (const float*)d_in[16];
  float* out = (float*)d_out;

  char* ws = (char*)d_ws;
  size_t off = 0;
  auto alloc = [&](size_t bytes) -> void* {
    void* p = ws + off;
    off += (bytes + 255) & ~(size_t)255;
    return p;
  };
  float*    h        = (float*)alloc((size_t)ROWS * D_MODEL * 4);
  float*    u_re     = (float*)alloc((size_t)ROWS * D_STATE * 4);
  float*    u_im     = (float*)alloc((size_t)ROWS * D_STATE * 4);
  unsigned* z_bf     = (unsigned*)alloc((size_t)ROWS * 32 * 4);
  unsigned* wf_embed = (unsigned*)alloc((size_t)2048 * 64 * 2);
  unsigned* wf_Bre   = (unsigned*)alloc((size_t)STAGES * 64 * 128 * 2);
  unsigned* wf_Bim   = (unsigned*)alloc((size_t)STAGES * 64 * 128 * 2);
  unsigned* wf_Cre   = (unsigned*)alloc((size_t)STAGES * 128 * 64 * 2);
  unsigned* wf_Cim   = (unsigned*)alloc((size_t)STAGES * 128 * 64 * 2);
  unsigned* wf_cls   = (unsigned*)alloc((size_t)64 * N_CLS_PAD * 2);

  auto gridFor = [](long total) { return (int)((total + 255) / 256); };
  // Pre-swizzle all weights into WMMA B-fragment layout (bf16).
  prep_frag<<<gridFor(65536), 256, 0, stream>>>(W_embed, wf_embed, 2048, 64, 64, 0, 64, 1, 0, 0);
  prep_frag<<<gridFor(20L * 4096), 256, 0, stream>>>(B_re, wf_Bre, 64, 128, 128, 1, 64,
                                                     STAGES, (long)D_STATE * D_MODEL, 4096);
  prep_frag<<<gridFor(20L * 4096), 256, 0, stream>>>(B_im, wf_Bim, 64, 128, 128, 1, 64,
                                                     STAGES, (long)D_STATE * D_MODEL, 4096);
  prep_frag<<<gridFor(20L * 4096), 256, 0, stream>>>(C_re, wf_Cre, 128, 64, 64, 1, 128,
                                                     STAGES, (long)D_MODEL * D_STATE, 4096);
  prep_frag<<<gridFor(20L * 4096), 256, 0, stream>>>(C_im, wf_Cim, 128, 64, 64, 1, 128,
                                                     STAGES, (long)D_MODEL * D_STATE, 4096);
  prep_frag<<<gridFor(6656), 256, 0, stream>>>(W_cls, wf_cls, 64, N_CLS_PAD, N_CLS, 0,
                                               N_CLS, 1, 0, 0);

  const int nTiles = ROWS / 16;   // 1024 row tiles
  k_embed<<<nTiles, 128, 0, stream>>>(x, ln0_g, ln0_b, wf_embed, b_embed, h);
  for (int s = 0; s < STAGES; ++s) {
    k_stage_pre<<<nTiles, 128, 0, stream>>>(
        h, ln_g + s * D_MODEL, ln_b + s * D_MODEL,
        wf_Bre + (size_t)s * 4096, wf_Bim + (size_t)s * 4096,
        gamma_log + s * D_STATE, u_re, u_im, z_bf);
    k_scan<<<BATCH * D_STATE, 128, 0, stream>>>(
        nu_log + s * D_STATE, theta_log + s * D_STATE, u_re, u_im);
    k_stage_post<<<nTiles, 128, 0, stream>>>(
        u_re, u_im, wf_Cre + (size_t)s * 4096, wf_Cim + (size_t)s * 4096,
        Dskip + s * D_MODEL, z_bf, h);
  }
  k_cls<<<nTiles, 128, 0, stream>>>(h, wf_cls, b_cls, out);
}
